// MultiHeadAttention2_86904368267445
// MI455X (gfx1250) — compile-verified, matrix operand order checked by reading, not run on hardware
//
#include <hip/hip_runtime.h>

#define B_  4
#define S_  2048
#define D_  512
#define H_  8
#define DH_ 64
#define NEGV (-1.0e6f)

typedef __bf16 bf16;
typedef __attribute__((ext_vector_type(16))) __bf16 v16bf;
typedef __attribute__((ext_vector_type(8)))  __bf16 v8bf;
typedef __attribute__((ext_vector_type(8)))  float  v8f;
typedef __attribute__((ext_vector_type(4)))  float  v4f;

__device__ inline bf16 f2bf(float f) { return (bf16)f; }

// ---------------------------------------------------------------------------
// Async 16-byte global->LDS copy (GLOBAL_LOAD_ASYNC_TO_LDS_B128).
// ---------------------------------------------------------------------------
#if __has_builtin(__builtin_amdgcn_global_load_async_to_lds_b128)
#define HAVE_ASYNC_LDS 1
#else
#define HAVE_ASYNC_LDS 0
#endif

typedef int v4si __attribute__((vector_size(16)));
typedef __attribute__((address_space(1))) v4si* as1_v4si;
typedef __attribute__((address_space(3))) v4si* as3_v4si;

__device__ inline void copy16_g2l(const bf16* g, bf16* l) {
#if HAVE_ASYNC_LDS
  __builtin_amdgcn_global_load_async_to_lds_b128(
      (as1_v4si)(g), (as3_v4si)(l), 0, 0);
#else
  *(v8bf*)l = *(const v8bf*)g;
#endif
}
__device__ inline void async_wait() {
#if HAVE_ASYNC_LDS
#if __has_builtin(__builtin_amdgcn_s_wait_asynccnt)
  __builtin_amdgcn_s_wait_asynccnt(0);
#else
  asm volatile("s_wait_asynccnt 0x0" ::: "memory");
#endif
#endif
}

// ---------------------------------------------------------------------------
// Tensor Data Mover: one DMA per staged 2D bf16 tile (D# per ISA ch.8).
// This toolchain has the 6-arg builtin:
//   (u32x4 g0, i32x8 g1, i32x4 g2, i32x4 g3, i32x8 g4, i32 cpol)
// ---------------------------------------------------------------------------
#if __has_builtin(__builtin_amdgcn_tensor_load_to_lds)
#define HAVE_TDM 1
#else
#define HAVE_TDM 0
#endif

typedef unsigned u32x4 __attribute__((vector_size(16)));
typedef int      i32x4 __attribute__((vector_size(16)));
typedef int      i32x8 __attribute__((vector_size(32)));

#if HAVE_TDM
__device__ inline void tdm_load_2d(const bf16* gsrc, bf16* ldst,
                                   int tile_w, int tile_h, int stride_elems) {
  unsigned long long ga = (unsigned long long)(uintptr_t)gsrc;
  u32x4 g0;
  g0[0] = 1u;                                    // count=1, no gather
  g0[1] = (unsigned)(uintptr_t)ldst;             // lds_addr (bytes)
  g0[2] = (unsigned)ga;                          // global_addr[31:0]
  g0[3] = (unsigned)((ga >> 32) & 0x1FFFFFFull)  // global_addr[56:32]
        | (2u << 30);                            // type = 2 ("image")
  i32x8 g1;
  g1[0] = 1 << 16;                 // workgroup_mask=0, data_size=2B
  g1[1] = 0;                       // tensor_dim0[15:0]=0 (dim0 = 1M)
  g1[2] = 0x10;                    // tensor_dim0[31:16]=0x10, tensor_dim1[15:0]=0
  g1[3] = (tile_w << 16) | 0x10;   // tensor_dim1[31:16]=0x10 (1M), tile_dim0
  g1[4] = tile_h & 0xffff;         // tile_dim1 (tile_dim2=0)
  g1[5] = stride_elems;            // tensor_dim0_stride[31:0]
  g1[6] = 0;                       // stride0[47:32], stride1[15:0]
  g1[7] = 0;
  i32x4 z4 = {0, 0, 0, 0};
  i32x8 z8 = {0, 0, 0, 0, 0, 0, 0, 0};
  __builtin_amdgcn_tensor_load_to_lds(g0, g1, z4, z4, z8, 0);
}
#endif

__device__ inline void tensor_wait() {
#if HAVE_TDM
#if __has_builtin(__builtin_amdgcn_s_wait_tensorcnt)
  __builtin_amdgcn_s_wait_tensorcnt(0);
#else
  asm volatile("s_wait_tensorcnt 0x0" ::: "memory");
#endif
#endif
}

// ---------------------------------------------------------------------------
// 16-lane (DPP16 row) reductions via row_ror — pure VALU.
// ---------------------------------------------------------------------------
template <int N>
__device__ inline float lanerot16(float x) {
#if __has_builtin(__builtin_amdgcn_update_dpp)
  int xi = __builtin_bit_cast(int, x);
  int r  = __builtin_amdgcn_update_dpp(xi, xi, 0x120 | N, 0xf, 0xf, false);
  return __builtin_bit_cast(float, r);
#else
  return __shfl_xor(x, N, 32);
#endif
}
__device__ inline float red_max16(float x) {
  x = fmaxf(x, lanerot16<1>(x));
  x = fmaxf(x, lanerot16<2>(x));
  x = fmaxf(x, lanerot16<4>(x));
  x = fmaxf(x, lanerot16<8>(x));
  return x;
}
__device__ inline float red_sum16(float x) {
  x += lanerot16<1>(x);
  x += lanerot16<2>(x);
  x += lanerot16<4>(x);
  x += lanerot16<8>(x);
  return x;
}

// ---------------------------------------------------------------------------
// Weight transpose+convert+scale: W fp32 [K][N] -> Wt bf16 [N][K], once.
// ---------------------------------------------------------------------------
__global__ __launch_bounds__(256) void wt_bf16(const float* __restrict__ W,
                                               bf16* __restrict__ Wt,
                                               float scale) {
  __shared__ float tile[32][33];
  const int k0 = blockIdx.x * 32, n0 = blockIdx.y * 32;
  const int tx = threadIdx.x & 31, ty = threadIdx.x >> 5;   // 32 x 8
  #pragma unroll
  for (int i = 0; i < 4; ++i)
    tile[ty + 8 * i][tx] = W[(size_t)(k0 + ty + 8 * i) * D_ + n0 + tx];
  __syncthreads();
  #pragma unroll
  for (int i = 0; i < 4; ++i)
    Wt[(size_t)(n0 + ty + 8 * i) * D_ + k0 + tx] = f2bf(tile[tx][ty + 8 * i] * scale);
}

// ---------------------------------------------------------------------------
// GEMM: C[8192,512] = A[8192,512] * W[512,512], W given transposed bf16.
// Block tile 128x128, 8 waves; wave = 16 rows x 128 cols; K-step 32.
// Tile staging via TDM (one DMA per tile) when available.
// ---------------------------------------------------------------------------
template <int A_FP32, int OUT_MODE>
__global__ __launch_bounds__(256) void gemm_wmma_bf16(
    const void* __restrict__ Av,
    const bf16* __restrict__ Wt,   // [N][K] bf16
    void* __restrict__ Outv)
{
  constexpr int Kd = 512, Nd = 512;
  __shared__ bf16 Al[128 * 32];    // [row][k]
  __shared__ bf16 Bl[128 * 32];    // [n][k]

  const int t    = threadIdx.x;
  const int w    = t >> 5;
  const int lane = t & 31;
  const int ln   = lane & 15;
  const int hl   = lane >> 4;
  const int m0   = blockIdx.x * 128;
  const int n0   = blockIdx.y * 128;

  v8f acc[8] = {};

  for (int k0 = 0; k0 < Kd; k0 += 32) {
    // --- stage A tile ---
    if (A_FP32) {
      const int r  = t >> 1;
      const int c0 = (t & 1) * 16;
      const float* Ap = (const float*)Av + (size_t)(m0 + r) * Kd + k0 + c0;
      v4f f0 = *(const v4f*)(Ap);
      v4f f1 = *(const v4f*)(Ap + 4);
      v4f f2 = *(const v4f*)(Ap + 8);
      v4f f3 = *(const v4f*)(Ap + 12);
      v8bf t0, t1;
      #pragma unroll
      for (int j = 0; j < 4; ++j) {
        t0[j]     = f2bf(f0[j]);
        t0[j + 4] = f2bf(f1[j]);
        t1[j]     = f2bf(f2[j]);
        t1[j + 4] = f2bf(f3[j]);
      }
      *(v8bf*)&Al[r * 32 + c0]     = t0;
      *(v8bf*)&Al[r * 32 + c0 + 8] = t1;
    } else {
#if HAVE_TDM
      if (w == 1)
        tdm_load_2d((const bf16*)Av + (size_t)m0 * Kd + k0, Al, 32, 128, Kd);
#else
      const int r  = t >> 1;
      const int c0 = (t & 1) * 16;
      const bf16* Ap = (const bf16*)Av + (size_t)(m0 + r) * Kd + k0 + c0;
      copy16_g2l(Ap,     &Al[r * 32 + c0]);
      copy16_g2l(Ap + 8, &Al[r * 32 + c0 + 8]);
#endif
    }
    // --- stage W tile ---
#if HAVE_TDM
    if (w == 0)
      tdm_load_2d(Wt + (size_t)n0 * Kd + k0, Bl, 32, 128, Kd);
#else
    {
      const int n  = t >> 1;
      const int c0 = (t & 1) * 16;
      const bf16* Wp = Wt + (size_t)(n0 + n) * Kd + k0 + c0;
      copy16_g2l(Wp,     &Bl[n * 32 + c0]);
      copy16_g2l(Wp + 8, &Bl[n * 32 + c0 + 8]);
    }
#endif
    tensor_wait();
    async_wait();
    __syncthreads();

    v16bf af;
    {
      const int row = w * 16 + ln;
      ((v8bf*)&af)[0] = *(const v8bf*)&Al[row * 32 + 8 * hl];
      ((v8bf*)&af)[1] = *(const v8bf*)&Al[row * 32 + 16 + 8 * hl];
    }
    #pragma unroll
    for (int nt = 0; nt < 8; ++nt) {
      v16bf bfg;
      const int n = nt * 16 + ln;
      ((v8bf*)&bfg)[0] = *(const v8bf*)&Bl[n * 32 + 8 * hl];
      ((v8bf*)&bfg)[1] = *(const v8bf*)&Bl[n * 32 + 16 + 8 * hl];
      acc[nt] = __builtin_amdgcn_wmma_f32_16x16x32_bf16(
          false, af, false, bfg, (short)0, acc[nt], false, false);
    }
    __syncthreads();
  }

  #pragma unroll
  for (int nt = 0; nt < 8; ++nt) {
    #pragma unroll
    for (int r = 0; r < 8; ++r) {
      const int gm = m0 + w * 16 + r + 8 * hl;
      const int gn = n0 + nt * 16 + ln;
      if (OUT_MODE == 0) {
        ((float*)Outv)[(size_t)gm * Nd + gn] = acc[nt][r];
      } else {
        const int b = gm >> 11, s = gm & (S_ - 1);
        const int h = gn >> 6,  dh = gn & (DH_ - 1);
        ((bf16*)Outv)[(((size_t)(b * H_ + h)) * S_ + s) * DH_ + dh] = f2bf(acc[nt][r]);
      }
    }
  }
}

// ---------------------------------------------------------------------------
// Flash attention: WG = 128 queries of one (b,h); wave = 16 queries.
// 64-key blocks; K tile staged by TDM, V transposed manually.
// ---------------------------------------------------------------------------
#define KB 64
__global__ __launch_bounds__(256) void attn_wmma(
    const bf16* __restrict__ Q, const bf16* __restrict__ Kin,
    const bf16* __restrict__ V, const long long* __restrict__ mask_num,
    bf16* __restrict__ Out)
{
  __shared__ bf16 Kl[KB * 64];        // [key][dh]       8 KB
  __shared__ bf16 Vt[64 * KB];        // [dh][key]       8 KB
  __shared__ bf16 Pl[8 * 16 * KB];    // per-wave P     16 KB

  const int t    = threadIdx.x;
  const int w    = t >> 5;
  const int lane = t & 31;
  const int ln   = lane & 15;
  const int hl   = lane >> 4;

  const int bh    = blockIdx.y;
  const int b     = bh >> 3;
  const int h     = bh & 7;
  const int maskv = (int)mask_num[b];

  const bf16* Qb = Q   + (size_t)bh * S_ * DH_;
  const bf16* Kb = Kin + (size_t)bh * S_ * DH_;
  const bf16* Vb = V   + (size_t)bh * S_ * DH_;

  const int q0 = blockIdx.x * 128 + w * 16;

  v16bf qa[2];
  {
    const bf16* qp = Qb + (size_t)(q0 + ln) * DH_;
    #pragma unroll
    for (int c = 0; c < 2; ++c) {
      ((v8bf*)&qa[c])[0] = *(const v8bf*)(qp + 32 * c + 8 * hl);
      ((v8bf*)&qa[c])[1] = *(const v8bf*)(qp + 32 * c + 16 + 8 * hl);
    }
  }

  v8f o[4] = {};
  float rmax[8], rsum[8];
  #pragma unroll
  for (int r = 0; r < 8; ++r) { rmax[r] = -3.0e38f; rsum[r] = 0.f; }

  for (int kb0 = 0; kb0 < S_; kb0 += KB) {
    // --- stage K: one TDM DMA (64x64 bf16, contiguous) ---
#if HAVE_TDM
    if (w == 0)
      tdm_load_2d(Kb + (size_t)kb0 * DH_, Kl, 64, 64, 64);
#else
    {
      const int row = t >> 2;
      const int c0  = (t & 3) * 16;
      const bf16* kp = Kb + (size_t)(kb0 + row) * DH_ + c0;
      copy16_g2l(kp,     &Kl[row * 64 + c0]);
      copy16_g2l(kp + 8, &Kl[row * 64 + c0 + 8]);
    }
#endif
    // --- stage V transposed: thread owns key rows {2p,2p+1} x 8 dh ---
    {
      const int p  = t >> 3;
      const int c0 = (t & 7) * 8;
      const bf16* vp = Vb + (size_t)(kb0 + 2 * p) * DH_ + c0;
      v8bf v0 = *(const v8bf*)(vp);
      v8bf v1 = *(const v8bf*)(vp + DH_);
      #pragma unroll
      for (int j = 0; j < 8; ++j) {
        unsigned pk = (unsigned)__builtin_bit_cast(unsigned short, v0[j])
                    | ((unsigned)__builtin_bit_cast(unsigned short, v1[j]) << 16);
        *(unsigned*)&Vt[(c0 + j) * KB + 2 * p] = pk;
      }
    }
    if (kb0 + KB < S_) {
      __builtin_prefetch(Kb + (size_t)(kb0 + KB + (t >> 2)) * DH_, 0, 1);
      __builtin_prefetch(Vb + (size_t)(kb0 + KB + (t >> 3) * 2) * DH_, 0, 1);
    }
    tensor_wait();
    async_wait();
    __syncthreads();

    // --- scores: 4 x 16-key tiles, K-dim = DH = 64 (2 WMMA each) ---
    v8f s[4];
    #pragma unroll
    for (int nt = 0; nt < 4; ++nt) {
      v8f sc = {};
      #pragma unroll
      for (int c = 0; c < 2; ++c) {
        v16bf kf;
        const int key = nt * 16 + ln;
        ((v8bf*)&kf)[0] = *(const v8bf*)&Kl[key * 64 + 32 * c + 8 * hl];
        ((v8bf*)&kf)[1] = *(const v8bf*)&Kl[key * 64 + 32 * c + 16 + 8 * hl];
        sc = __builtin_amdgcn_wmma_f32_16x16x32_bf16(
            false, qa[c], false, kf, (short)0, sc, false, false);
      }
      const int keyg = kb0 + nt * 16 + ln;
      #pragma unroll
      for (int r = 0; r < 8; ++r)
        sc[r] = (keyg == maskv) ? NEGV : sc[r];   // scale pre-folded into Q
      s[nt] = sc;
    }

    // --- online softmax; row r+8*hl stats via DPP row reductions ---
    float nm[8], corr[8], psum[8];
    #pragma unroll
    for (int r = 0; r < 8; ++r) {
      float m01 = fmaxf(s[0][r], s[1][r]);
      float m23 = fmaxf(s[2][r], s[3][r]);
      nm[r] = red_max16(fmaxf(m01, m23));
      nm[r] = fmaxf(nm[r], rmax[r]);
      corr[r] = __expf(rmax[r] - nm[r]);
      rmax[r] = nm[r];
    }
    #pragma unroll
    for (int nt = 0; nt < 4; ++nt) {
      #pragma unroll
      for (int r = 0; r < 8; ++r) s[nt][r] = __expf(s[nt][r] - nm[r]);
    }
    #pragma unroll
    for (int r = 0; r < 8; ++r) {
      psum[r] = red_sum16((s[0][r] + s[1][r]) + (s[2][r] + s[3][r]));
      rsum[r] = rsum[r] * corr[r] + psum[r];
    }
    #pragma unroll
    for (int nt = 0; nt < 4; ++nt) {
      #pragma unroll
      for (int r = 0; r < 8; ++r) o[nt][r] *= corr[r];
    }

    // --- transpose P through wave-private LDS (same-wave DS in-order) ---
    bf16* Pw = &Pl[w * 16 * KB];
    #pragma unroll
    for (int nt = 0; nt < 4; ++nt) {
      #pragma unroll
      for (int r = 0; r < 8; ++r)
        Pw[(r + 8 * hl) * KB + nt * 16 + ln] = f2bf(s[nt][r]);
    }
    v16bf pf[2];
    #pragma unroll
    for (int c = 0; c < 2; ++c) {
      ((v8bf*)&pf[c])[0] = *(const v8bf*)&Pw[ln * KB + 32 * c + 8 * hl];
      ((v8bf*)&pf[c])[1] = *(const v8bf*)&Pw[ln * KB + 32 * c + 16 + 8 * hl];
    }

    // --- O += P * V (K-dim = 64 keys -> 2 WMMA per dh tile) ---
    #pragma unroll
    for (int nt = 0; nt < 4; ++nt) {
      const int dh = nt * 16 + ln;
      #pragma unroll
      for (int c = 0; c < 2; ++c) {
        v16bf vf;
        ((v8bf*)&vf)[0] = *(const v8bf*)&Vt[dh * KB + 32 * c + 8 * hl];
        ((v8bf*)&vf)[1] = *(const v8bf*)&Vt[dh * KB + 32 * c + 16 + 8 * hl];
        o[nt] = __builtin_amdgcn_wmma_f32_16x16x32_bf16(
            false, pf[c], false, vf, (short)0, o[nt], false, false);
      }
    }
    __syncthreads();
  }

  float inv[8];
  #pragma unroll
  for (int r = 0; r < 8; ++r) inv[r] = 1.0f / rsum[r];
  #pragma unroll
  for (int nt = 0; nt < 4; ++nt) {
    #pragma unroll
    for (int r = 0; r < 8; ++r) {
      const float val = o[nt][r] * inv[r];
      const int qg  = q0 + r + 8 * hl;
      const int col = h * DH_ + nt * 16 + ln;
      Out[((size_t)b * S_ + qg) * D_ + col] = f2bf(val);
    }
  }
}

// ---------------------------------------------------------------------------
extern "C" void kernel_launch(void* const* d_in, const int* in_sizes, int n_in,
                              void* d_out, int out_size, void* d_ws, size_t ws_size,
                              hipStream_t stream) {
  (void)in_sizes; (void)n_in; (void)out_size; (void)ws_size;

  const float* queries = (const float*)d_in[0];
  const float* keys    = (const float*)d_in[1];
  const float* values  = (const float*)d_in[2];
  const float* Wq = (const float*)d_in[3];
  const float* Wk = (const float*)d_in[4];
  const float* Wv = (const float*)d_in[5];
  const float* Wo = (const float*)d_in[6];
  const long long* mask_num = (const long long*)d_in[7];  // int64

  const size_t NELEM = (size_t)B_ * S_ * D_;   // 4M elements
  const size_t WELEM = (size_t)D_ * D_;        // 256K elements
  bf16* Wtq = (bf16*)d_ws;
  bf16* Wtk = Wtq + WELEM;
  bf16* Wtv = Wtk + WELEM;
  bf16* Wto = Wtv + WELEM;
  bf16* Qb  = Wto + WELEM;        // [B][H][S][DH]
  bf16* Kb  = Qb + NELEM;
  bf16* Vb  = Kb + NELEM;
  bf16* Ab  = Vb + NELEM;         // [B][S][H*DH]

  dim3 wtg(D_ / 32, D_ / 32);
  wt_bf16<<<wtg, 256, 0, stream>>>(Wq, Wtq, 0.125f);   // fold 1/sqrt(DH)
  wt_bf16<<<wtg, 256, 0, stream>>>(Wk, Wtk, 1.0f);
  wt_bf16<<<wtg, 256, 0, stream>>>(Wv, Wtv, 1.0f);
  wt_bf16<<<wtg, 256, 0, stream>>>(Wo, Wto, 1.0f);

  dim3 gg((B_ * S_) / 128, D_ / 128);
  gemm_wmma_bf16<1, 1><<<gg, 256, 0, stream>>>(queries, Wtq, Qb);
  gemm_wmma_bf16<1, 1><<<gg, 256, 0, stream>>>(keys,    Wtk, Kb);
  gemm_wmma_bf16<1, 1><<<gg, 256, 0, stream>>>(values,  Wtv, Vb);

  attn_wmma<<<dim3(S_ / 128, B_ * H_), 256, 0, stream>>>(Qb, Kb, Vb, mask_num, Ab);

  gemm_wmma_bf16<0, 0><<<gg, 256, 0, stream>>>(Ab, Wto, d_out);
}
